// Attention_21277267984815
// MI455X (gfx1250) — matrix operand order repaired and checked by reading, not compiled, not run on hardware
//
#include <hip/hip_runtime.h>

#define S 8192
#define H 1024

typedef __attribute__((ext_vector_type(16))) _Float16 v16h;
typedef __attribute__((ext_vector_type(8)))  _Float16 v8h;
typedef __attribute__((ext_vector_type(8)))  float    v8f;
typedef __attribute__((ext_vector_type(4)))  unsigned u32x4;
typedef __attribute__((ext_vector_type(4)))  int      i32x4;
typedef __attribute__((ext_vector_type(8)))  int      i32x8;

static __device__ __forceinline__ v16h combine16(v8h lo, v8h hi) {
    v16h r;
#pragma unroll
    for (int i = 0; i < 8; ++i) { r[i] = lo[i]; r[8 + i] = hi[i]; }
    return r;
}

// A-matrix 16x32 f16 tile from row-major [M][K] memory.
// lane<16: M=lane, K=0..15 contiguous ; lane>=16: M=lane-16, K=16..31 contiguous
static __device__ __forceinline__ v16h load_a_f16(const _Float16* base, int ld, int lane) {
    const int m    = lane & 15;
    const int koff = (lane & 16);   // 0 or 16
    const _Float16* p = base + m * ld + koff;
    return combine16(*(const v8h*)p, *(const v8h*)(p + 8));
}

// B-matrix 32x16 f16 tile; memory holds Bᵀ row-major ([N][K], ld elems):
// lane<16: N=lane, K=0..15 contiguous ; lane>=16: N=lane-16, K=16..31 contiguous
static __device__ __forceinline__ v16h load_b_f16(const _Float16* base, int ld, int lane) {
    const int n    = lane & 15;
    const int koff = (lane & 16);   // 0 or 16
    const _Float16* p = base + n * ld + koff;
    return combine16(*(const v8h*)p, *(const v8h*)(p + 8));
}

static __device__ __forceinline__ v8f wmma_f16(v16h a, v16h b, v8f c) {
    return __builtin_amdgcn_wmma_f32_16x16x32_f16(false, a, false, b, (short)0, c,
                                                  false, false);
}

// -------- one-time f32 -> f16 conversion (8 elems/thread, b128 traffic) --------
__global__ __launch_bounds__(256)
void cvt_f32_to_f16(const float* __restrict__ in, _Float16* __restrict__ out) {
    const size_t i = ((size_t)blockIdx.x * 256 + threadIdx.x) * 8;
#pragma unroll
    for (int e = 0; e < 8; ++e) out[i + e] = (_Float16)in[i + e];
}

// -------- projection: out[s,o] = (tanh?) sum_h enc[s,h] * w[o,h], f16 in/out ------
// one wave -> 16x64 tile; grid 1024 blocks x 256 threads = 8192 wave-tiles (512 x 16)
__global__ __launch_bounds__(256)
void proj_kernel(const _Float16* __restrict__ enc, const _Float16* __restrict__ w,
                 _Float16* __restrict__ out, int do_tanh) {
    const int lane = threadIdx.x & 31;
    const int gw   = blockIdx.x * 8 + (threadIdx.x >> 5);
    const int row0 = (gw >> 4) << 4;   // 0..8176
    const int col0 = (gw & 15) << 6;   // 0..960

    v8f c0 = {}, c1 = {}, c2 = {}, c3 = {};
#pragma unroll 2
    for (int kk = 0; kk < H; kk += 32) {
        v16h a  = load_a_f16(enc + row0 * H + kk, H, lane);
        v16h b0 = load_b_f16(w + (col0 +  0) * H + kk, H, lane);
        v16h b1 = load_b_f16(w + (col0 + 16) * H + kk, H, lane);
        v16h b2 = load_b_f16(w + (col0 + 32) * H + kk, H, lane);
        v16h b3 = load_b_f16(w + (col0 + 48) * H + kk, H, lane);
        c0 = wmma_f16(a, b0, c0);
        c1 = wmma_f16(a, b1, c1);
        c2 = wmma_f16(a, b2, c2);
        c3 = wmma_f16(a, b3, c3);
    }
    const int mrow = (lane & 16) >> 1;  // 0 or 8
    const int ncol = lane & 15;
#pragma unroll
    for (int v = 0; v < 8; ++v) {
        const int r = row0 + mrow + v;
        float x0 = c0[v], x1 = c1[v], x2 = c2[v], x3 = c3[v];
        if (do_tanh) { x0 = tanhf(x0); x1 = tanhf(x1); x2 = tanhf(x2); x3 = tanhf(x3); }
        out[r * H + col0 +  0 + ncol] = (_Float16)x0;
        out[r * H + col0 + 16 + ncol] = (_Float16)x1;
        out[r * H + col0 + 32 + ncol] = (_Float16)x2;
        out[r * H + col0 + 48 + ncol] = (_Float16)x3;
    }
}

// -------- scores[i,j] = sum_h q[i,h] * k[j,h]  (raw logits into attn region) ------
// block = 8 waves -> 128x64 block tile; B (k-tile) staged into LDS by the
// Tensor Data Mover (tensor_load_to_lds + s_wait_tensorcnt), shared by all waves.
__global__ __launch_bounds__(256)
void scores_kernel(const _Float16* __restrict__ q, const _Float16* __restrict__ k,
                   float* __restrict__ attn) {
    __shared__ __align__(16) _Float16 Bs[64 * 32];   // 4 KB: 64 cols x 32 K
    const int lane = threadIdx.x & 31;
    const int wave = threadIdx.x >> 5;
    const int row0 = blockIdx.y * 128 + wave * 16;
    const int col0 = blockIdx.x * 64;

    const int n    = lane & 15;
    const int koff = (lane & 16);            // 0 or 16

    // Loop-invariant D# pieces (ISA 08_async_tensor §8).
    // group1: data_size=2B (code 1), no multicast/iterate/pad,
    //         tensor_dim0=H, tensor_dim1=S, tile_dim0=32, tile_dim1=64,
    //         tensor_dim0_stride=H.
    i32x8 g1;
    g1[0] = (1 << 16);          // data_size=1 (2 bytes)
    g1[1] = (H << 16);          // tensor_dim0[15:0] in bits [63:48]
    g1[2] = (S << 16);          // tensor_dim0[31:16]=0 | tensor_dim1[15:0]
    g1[3] = (32 << 16);         // tensor_dim1[31:16]=0 | tile_dim0=32
    g1[4] = 64;                 // tile_dim1=64, tile_dim2=0
    g1[5] = H;                  // tensor_dim0_stride[31:0]
    g1[6] = 0;                  // stride[47:32]=0, dim1_stride low=0
    g1[7] = 0;
    const i32x4 gz  = {0, 0, 0, 0};              // groups 2/3 unused (2-D tile)
    const i32x8 gz8 = {0, 0, 0, 0, 0, 0, 0, 0};  // extra descriptor words (unused)
    const unsigned long long ga0 =
        (unsigned long long)(size_t)(k + (size_t)col0 * H);

    v8f c0 = {}, c1 = {}, c2 = {}, c3 = {};
#pragma unroll 1
    for (int kk = 0; kk < H; kk += 32) {
        __syncthreads();                     // previous tile fully consumed
        if (wave == 0) {
            const unsigned long long ga = ga0 + (unsigned long long)kk * 2;
            u32x4 g0;
            g0[0] = 1u;                                          // count=1 (valid D#)
            g0[1] = 0u;                                          // lds_addr: Bs is the
                                                                 // only static LDS obj
            g0[2] = (unsigned)ga;                                // global_addr[31:0]
            g0[3] = (unsigned)((ga >> 32) & 0x1FFFFFFull)        // global_addr[56:32]
                  | (2u << 30);                                  // type=2 ("image")
            __builtin_amdgcn_tensor_load_to_lds(g0, g1, gz, gz, gz8, 0);
            __builtin_amdgcn_s_wait_tensorcnt(0);
        }
        __syncthreads();                     // TDM data visible to all waves
        // The TDM write is invisible to alias analysis (it targets the LDS
        // aperture via a descriptor, not the Bs symbol). Escape Bs and clobber
        // memory so the compiler keeps (and re-issues) the ds_load_b128s.
        asm volatile("" : : "v"((const _Float16*)Bs) : "memory");

        v16h a  = load_a_f16(q + row0 * H + kk, H, lane);
        v16h b0 = combine16(*(const v8h*)&Bs[( 0 + n) * 32 + koff],
                            *(const v8h*)&Bs[( 0 + n) * 32 + koff + 8]);
        v16h b1 = combine16(*(const v8h*)&Bs[(16 + n) * 32 + koff],
                            *(const v8h*)&Bs[(16 + n) * 32 + koff + 8]);
        v16h b2 = combine16(*(const v8h*)&Bs[(32 + n) * 32 + koff],
                            *(const v8h*)&Bs[(32 + n) * 32 + koff + 8]);
        v16h b3 = combine16(*(const v8h*)&Bs[(48 + n) * 32 + koff],
                            *(const v8h*)&Bs[(48 + n) * 32 + koff + 8]);
        c0 = wmma_f16(a, b0, c0);
        c1 = wmma_f16(a, b1, c1);
        c2 = wmma_f16(a, b2, c2);
        c3 = wmma_f16(a, b3, c3);
    }
    const int mrow = (lane & 16) >> 1;
    const int ncol = lane & 15;
#pragma unroll
    for (int v = 0; v < 8; ++v) {
        const size_t r = (size_t)(row0 + mrow + v) * S;
        attn[r + col0 +  0 + ncol] = c0[v];
        attn[r + col0 + 16 + ncol] = c1[v];
        attn[r + col0 + 32 + ncol] = c2[v];
        attn[r + col0 + 48 + ncol] = c3[v];
    }
}

// -------- softmax over axis i (rows) for each column j, in place; also col sums ----
__global__ __launch_bounds__(256)
void softmax_kernel(float* __restrict__ attn, float* __restrict__ colsum) {
    const int j = blockIdx.x * 256 + threadIdx.x;
    float m = -3.402823466e38f, l = 0.0f;
#pragma unroll 4
    for (int i = 0; i < S; ++i) {
        const float s  = attn[(size_t)i * S + j];
        const float nm = fmaxf(m, s);
        l = l * __expf(m - nm) + __expf(s - nm);
        m = nm;
    }
    const float inv = 1.0f / l;
    float cs = 0.0f;
#pragma unroll 4
    for (int i = 0; i < S; ++i) {
        const size_t idx = (size_t)i * S + j;
        const float v = __expf(attn[idx] - m) * inv;
        attn[idx] = v;
        cs += v;
    }
    colsum[j] = cs;   // mathematically ~1.0; kept for fidelity with reference
}

// -------- context[j,h] = colsum[j] * enc[j,h] --------
__global__ __launch_bounds__(256)
void context_kernel(const float* __restrict__ enc, const float* __restrict__ colsum,
                    float* __restrict__ ctx) {
    const int idx = blockIdx.x * 256 + threadIdx.x;   // S*H elements
    ctx[idx] = enc[idx] * colsum[idx >> 10];          // H == 1024
}

extern "C" void kernel_launch(void* const* d_in, const int* in_sizes, int n_in,
                              void* d_out, int out_size, void* d_ws, size_t ws_size,
                              hipStream_t stream) {
    const float* enc = (const float*)d_in[0];   // [1, S, H]
    const float* w1  = (const float*)d_in[1];   // [H, H] (out, in)
    const float* w2  = (const float*)d_in[2];   // [H, H]

    float* ctx  = (float*)d_out;                       // context [1,S,H]
    float* attn = (float*)d_out + (size_t)S * H;       // attn    [1,S,S]

    _Float16* qh   = (_Float16*)d_ws;                  // q f16   [S,H]  16 MB
    _Float16* kh   = qh   + (size_t)S * H;             // k f16   [S,H]  16 MB
    _Float16* ench = kh   + (size_t)S * H;             // enc f16 [S,H]  16 MB
    _Float16* w1h  = ench + (size_t)S * H;             // w1 f16  [H,H]   2 MB
    _Float16* w2h  = w1h  + (size_t)H * H;             // w2 f16  [H,H]   2 MB
    float*    csum = (float*)(w2h + (size_t)H * H);    // [S] col sums   32 KB

    cvt_f32_to_f16<<<(S * H) / 2048, 256, 0, stream>>>(enc, ench);
    cvt_f32_to_f16<<<(H * H) / 2048, 256, 0, stream>>>(w1, w1h);
    cvt_f32_to_f16<<<(H * H) / 2048, 256, 0, stream>>>(w2, w2h);

    proj_kernel<<<1024, 256, 0, stream>>>(ench, w1h, qh, 1);  // q = tanh(enc @ w1ᵀ)
    proj_kernel<<<1024, 256, 0, stream>>>(ench, w2h, kh, 0);  // k = enc @ w2ᵀ

    dim3 grid_sc(S / 64, S / 128);                            // 128 x 64 blocks
    scores_kernel<<<grid_sc, 256, 0, stream>>>(qh, kh, attn);

    softmax_kernel<<<S / 256, 256, 0, stream>>>(attn, csum);

    context_kernel<<<(S * H) / 256, 256, 0, stream>>>(enc, csum, ctx);
}
